// Loss_47321949667996
// MI455X (gfx1250) — matrix operand order repaired and checked by reading, not compiled, not run on hardware
//
#include <hip/hip_runtime.h>
#include <hip/hip_bf16.h>
#include <stdint.h>

// ---------------------------------------------------------------------------
// NT-Xent loss for MI455X (gfx1250, wave32, WMMA).
//
//   B=4096, D=512, N=8192.  sim = (h h^T) * 2   (T = 0.5)
//   loss = sum_i( logsumexp_{j!=i}(sim[i,j]) - sim[i, i^4096] ) / N
//
// Fused GEMM + online logsumexp; fp32 split into bf16 hi/lo planes, each fp32
// MAC emulated by 3x v_wmma_f32_16x16x32_bf16 (hi*hi + hi*lo + lo*hi).
// Global->LDS staging uses GLOBAL_LOAD_ASYNC_TO_LDS_B128 (ASYNCcnt) with
// double-buffered LDS so loads overlap WMMA and the stats phase.
// ---------------------------------------------------------------------------

typedef __attribute__((ext_vector_type(16))) __bf16 v16bf;
typedef __attribute__((ext_vector_type(8)))  float  v8f;
typedef int v4i __attribute__((vector_size(16)));

#define N_TOT          8192
#define D_DIM          512
#define B_HALF         4096
#define ROWS_PER_BLK   128
#define COLS_PER_TILE  128
#define COL_SPLITS     4
#define COLS_PER_BLK   (N_TOT / COL_SPLITS)           // 2048
#define TILES_PER_BLK  (COLS_PER_BLK / COLS_PER_TILE) // 16
#define NKCH           (D_DIM / 32)                   // 16 k-chunks of 32
#define TOTALCH        (TILES_PER_BLK * NKCH)         // 256 chunks per block

#if defined(__has_builtin)
#if __has_builtin(__builtin_amdgcn_global_load_async_to_lds_b128) && \
    __has_builtin(__builtin_amdgcn_s_wait_asynccnt)
#define USE_ASYNC_LDS 1
#endif
#endif

// builtin signature: (v4i addrspace(1)*, v4i addrspace(3)*, imm offset, imm cpol)
#define AS1_V4(p) ((__attribute__((address_space(1))) v4i*)(p))
#define AS3_V4(p) ((__attribute__((address_space(3))) v4i*)(p))

#ifdef USE_ASYNC_LDS
#define WAIT_ASYNC_8() __builtin_amdgcn_s_wait_asynccnt(8)
#define WAIT_ASYNC_0() __builtin_amdgcn_s_wait_asynccnt(0)
#else
#define WAIT_ASYNC_8()
#define WAIT_ASYNC_0()
#endif

__device__ __forceinline__ unsigned short f2bf(float f) {
  unsigned u = __float_as_uint(f);
  unsigned r = u + 0x7FFFu + ((u >> 16) & 1u);   // round-to-nearest-even
  return (unsigned short)(r >> 16);
}
__device__ __forceinline__ float bf2f(unsigned short h) {
  return __uint_as_float(((unsigned)h) << 16);
}

// ---------------------------------------------------------------------------
// Kernel 1: concat(h_i,h_j) -> bf16 hi/lo planes; also zero the output scalar.
// ---------------------------------------------------------------------------
__global__ void cvt_split_kernel(const float* __restrict__ h_i,
                                 const float* __restrict__ h_j,
                                 unsigned short* __restrict__ ghi,
                                 unsigned short* __restrict__ glo,
                                 float* __restrict__ out) {
  if (blockIdx.x == 0 && threadIdx.x == 0) out[0] = 0.0f;
  const int BD = B_HALF * D_DIM;
  int idx0 = (blockIdx.x * blockDim.x + threadIdx.x) * 4;
  const float* src = (idx0 < BD) ? (h_i + idx0) : (h_j + (idx0 - BD));
  float4 a = *(const float4*)src;
  float v[4] = {a.x, a.y, a.z, a.w};
  unsigned short h[4], l[4];
#pragma unroll
  for (int e = 0; e < 4; ++e) {
    h[e] = f2bf(v[e]);
    l[e] = f2bf(v[e] - bf2f(h[e]));
  }
  *(ushort4*)(ghi + idx0) = make_ushort4(h[0], h[1], h[2], h[3]);
  *(ushort4*)(glo + idx0) = make_ushort4(l[0], l[1], l[2], l[3]);
}

// ---------------------------------------------------------------------------
// Stage one 32-wide k-chunk of the A(128 rows) and B(128 cols) hi/lo panels
// into one LDS buffer (plane layout: [Ahi|Alo|Bhi|Blo] x 512 uint4).
// ---------------------------------------------------------------------------
__device__ __forceinline__ void stage_chunk(const uint4* __restrict__ ghi,
                                            const uint4* __restrict__ glo,
                                            uint4* sdst, int rowbase, int j0,
                                            int kc, int t) {
#pragma unroll
  for (int s = t; s < 512; s += 256) {
    const int row = s >> 2, part = s & 3;            // 64 uint4 per D-row
    const int ga = (rowbase + row) * 64 + kc * 4 + part;
    const int gb = (j0 + row) * 64 + kc * 4 + part;
#ifdef USE_ASYNC_LDS
    __builtin_amdgcn_global_load_async_to_lds_b128(AS1_V4(ghi + ga), AS3_V4(sdst + s),        0, 0);
    __builtin_amdgcn_global_load_async_to_lds_b128(AS1_V4(glo + ga), AS3_V4(sdst + 512 + s),  0, 0);
    __builtin_amdgcn_global_load_async_to_lds_b128(AS1_V4(ghi + gb), AS3_V4(sdst + 1024 + s), 0, 0);
    __builtin_amdgcn_global_load_async_to_lds_b128(AS1_V4(glo + gb), AS3_V4(sdst + 1536 + s), 0, 0);
#else
    sdst[s]        = ghi[ga];
    sdst[512 + s]  = glo[ga];
    sdst[1024 + s] = ghi[gb];
    sdst[1536 + s] = glo[gb];
#endif
  }
}

// ---------------------------------------------------------------------------
// Kernel 2: fused GEMM + streaming logsumexp.
// Block: 256 threads = 8 waves; wave w owns rows [rowbase+16w, +16).
// Flat software pipeline over (tile, kchunk) with double-buffered async LDS.
// ---------------------------------------------------------------------------
__global__ __launch_bounds__(256) void ntxent_gemm_lse_kernel(
    const uint4* __restrict__ ghi, const uint4* __restrict__ glo,
    float* __restrict__ mPart, float* __restrict__ lPart,
    float* __restrict__ pPart) {

  __shared__ uint4 sbuf[2][2048];                 // 2 x 32KB double buffer

  const int t      = threadIdx.x;
  const int wave   = t >> 5;
  const int lane   = t & 31;
  const int l15    = lane & 15;
  const int hiHalf = lane >> 4;                   // 0: lanes 0-15, 1: 16-31

  const int rg       = blockIdx.x >> 2;           // row group (0..63)
  const int cs       = blockIdx.x & (COL_SPLITS - 1);
  const int rowbase  = rg * ROWS_PER_BLK;
  const int waveRow0 = rowbase + (wave << 4);
  // the 16 pos-columns of this wave's rows all land in exactly one col tile:
  const int posTile  = (waveRow0 ^ B_HALF) & ~(COLS_PER_TILE - 1);
  const int colBase  = cs * COLS_PER_BLK;

  float runmax[8], runsum[8], posv[8];
#pragma unroll
  for (int r = 0; r < 8; ++r) { runmax[r] = -1e30f; runsum[r] = 0.f; posv[r] = 0.f; }

  v8f acc[8];
  const v8f vzero = {0.f, 0.f, 0.f, 0.f, 0.f, 0.f, 0.f, 0.f};
#pragma unroll
  for (int n = 0; n < 8; ++n) acc[n] = vzero;

  union Frag { v16bf v; uint4 q[2]; };

  // pipeline prologue: chunk 0 -> buffer 0
  stage_chunk(ghi, glo, &sbuf[0][0], rowbase, colBase, 0, t);

  for (int c = 0; c < TOTALCH; ++c) {
    const int buf  = c & 1;
    const int tile = c >> 4;
    const int j0   = colBase + tile * COLS_PER_TILE;

    if (c + 1 < TOTALCH) {
      const int nt = (c + 1) >> 4;
      stage_chunk(ghi, glo, &sbuf[1 - buf][0], rowbase,
                  colBase + nt * COLS_PER_TILE, (c + 1) & 15, t);
      WAIT_ASYNC_8();   // oldest 8 ops (chunk c) landed; chunk c+1 in flight
    } else {
      WAIT_ASYNC_0();
    }
    __syncthreads();    // chunk c visible to every wave

    // --- A fragments (ISA 16-bit A 16x32 layout):
    // lanes 0-15:  K0-7  @byte 0,  K16-23 @byte 32
    // lanes 16-31: K8-15 @byte 16, K24-31 @byte 48
    const uint4* sA_hi = &sbuf[buf][0];
    const uint4* sA_lo = &sbuf[buf][512];
    const uint4* sB_hi = &sbuf[buf][1024];
    const uint4* sB_lo = &sbuf[buf][1536];

    Frag ahi, alo;
    {
      const int arow = (wave << 4) + l15;
      const uint4* pa = &sA_hi[arow * 4];
      const uint4* pl = &sA_lo[arow * 4];
      ahi.q[0] = pa[hiHalf];  ahi.q[1] = pa[hiHalf + 2];
      alo.q[0] = pl[hiHalf];  alo.q[1] = pl[hiHalf + 2];
    }

    // --- B fragments (32x16): lanes 0-15 = K0-7,K16-23; lanes 16-31 = K8-15,K24-31
#pragma unroll
    for (int n = 0; n < 8; ++n) {
      Frag bhi, blo;
      const int bcol = (n << 4) + l15;
      const int off  = hiHalf;
      const uint4* pb = &sB_hi[bcol * 4];
      const uint4* ql = &sB_lo[bcol * 4];
      bhi.q[0] = pb[off]; bhi.q[1] = pb[off + 2];
      blo.q[0] = ql[off]; blo.q[1] = ql[off + 2];

      acc[n] = __builtin_amdgcn_wmma_f32_16x16x32_bf16(
          false, ahi.v, false, bhi.v, (short)0, acc[n], false, false);
      acc[n] = __builtin_amdgcn_wmma_f32_16x16x32_bf16(
          false, ahi.v, false, blo.v, (short)0, acc[n], false, false);
      acc[n] = __builtin_amdgcn_wmma_f32_16x16x32_bf16(
          false, alo.v, false, bhi.v, (short)0, acc[n], false, false);
    }
    __syncthreads();    // all waves done reading sbuf[buf]

    if ((c & 15) == 15) {
      // --- online logsumexp update over this 16x128 sim tile ---------------
      // C layout: VGPR r, lanes 0-15 -> row r, lanes 16-31 -> row r+8;
      // column = j0 + 16n + (lane&15). Next tile's chunk 0 loads are already
      // in flight, hiding their latency behind this VALU phase.
      const bool hasPos = (j0 == posTile);
#pragma unroll
      for (int r = 0; r < 8; ++r) {
        const int rowg = waveRow0 + r + hiHalf * 8;
        float vv[8];
        float tmax = -1e30f;
        float pv   = 0.f;
#pragma unroll
        for (int n = 0; n < 8; ++n) {
          const int colg = j0 + (n << 4) + l15;
          float v = 2.0f * acc[n][r];               // 1/T = 2
          if (colg == rowg) v = -1e30f;             // mask self-similarity
          if (hasPos && colg == (rowg ^ B_HALF)) pv = v;  // positive pair
          vv[n] = v;
          tmax  = fmaxf(tmax, v);
        }
#pragma unroll
        for (int m = 1; m < 16; m <<= 1) tmax = fmaxf(tmax, __shfl_xor(tmax, m, 32));
        const float nm = fmaxf(runmax[r], tmax);
        float s = 0.f;
#pragma unroll
        for (int n = 0; n < 8; ++n) s += __expf(vv[n] - nm);
#pragma unroll
        for (int m = 1; m < 16; m <<= 1) s += __shfl_xor(s, m, 32);
        runsum[r] = runsum[r] * __expf(runmax[r] - nm) + s;
        runmax[r] = nm;
        if (hasPos) {
#pragma unroll
          for (int m = 1; m < 16; m <<= 1) pv += __shfl_xor(pv, m, 32);
          posv[r] += pv;
        }
      }
#pragma unroll
      for (int n = 0; n < 8; ++n) acc[n] = vzero;
    }
  }

  // --- write per-row partials (stats are replicated within each lane-half) --
  if (l15 == 0) {
#pragma unroll
    for (int r = 0; r < 8; ++r) {
      const int row = waveRow0 + r + hiHalf * 8;
      mPart[row * COL_SPLITS + cs] = runmax[r];
      lPart[row * COL_SPLITS + cs] = runsum[r];
      pPart[row * COL_SPLITS + cs] = posv[r];
    }
  }
}

// ---------------------------------------------------------------------------
// Kernel 3: combine column-split partials per row, reduce to the scalar loss.
// ---------------------------------------------------------------------------
__global__ void ntxent_reduce_kernel(const float* __restrict__ mPart,
                                     const float* __restrict__ lPart,
                                     const float* __restrict__ pPart,
                                     float* __restrict__ out) {
  const int row = blockIdx.x * blockDim.x + threadIdx.x;
  float m = -1e30f;
#pragma unroll
  for (int c = 0; c < COL_SPLITS; ++c) m = fmaxf(m, mPart[row * COL_SPLITS + c]);
  float l = 0.f, p = 0.f;
#pragma unroll
  for (int c = 0; c < COL_SPLITS; ++c) {
    l += lPart[row * COL_SPLITS + c] * __expf(mPart[row * COL_SPLITS + c] - m);
    p += pPart[row * COL_SPLITS + c];
  }
  float val = (logf(l) + m) - p;      // lse - pos
#pragma unroll
  for (int s = 1; s < 32; s <<= 1) val += __shfl_xor(val, s, 32);
  if ((threadIdx.x & 31) == 0) atomicAdd(out, val * (1.0f / (float)N_TOT));
}

// ---------------------------------------------------------------------------
extern "C" void kernel_launch(void* const* d_in, const int* in_sizes, int n_in,
                              void* d_out, int out_size, void* d_ws, size_t ws_size,
                              hipStream_t stream) {
  (void)in_sizes; (void)n_in; (void)out_size; (void)ws_size;
  const float* h_i = (const float*)d_in[0];
  const float* h_j = (const float*)d_in[1];
  float* out = (float*)d_out;

  // workspace: bf16 hi plane (8MB) | bf16 lo plane (8MB) | m/l/pos partials
  unsigned short* ghi = (unsigned short*)d_ws;
  unsigned short* glo = ghi + (size_t)N_TOT * D_DIM;
  float* mPart = (float*)(glo + (size_t)N_TOT * D_DIM);
  float* lPart = mPart + N_TOT * COL_SPLITS;
  float* pPart = lPart + N_TOT * COL_SPLITS;

  const int totalVec4 = (N_TOT * D_DIM) / 4;                  // 1,048,576
  cvt_split_kernel<<<totalVec4 / 256, 256, 0, stream>>>(h_i, h_j, ghi, glo, out);

  const int grid = (N_TOT / ROWS_PER_BLK) * COL_SPLITS;       // 256 blocks
  ntxent_gemm_lse_kernel<<<grid, 256, 0, stream>>>(
      (const uint4*)ghi, (const uint4*)glo, mPart, lPart, pPart);

  ntxent_reduce_kernel<<<N_TOT / 256, 256, 0, stream>>>(mPart, lPart, pPart, out);
}
